// STGCNBlock_8993661517873
// MI455X (gfx1250) — matrix operand order repaired and checked by reading, not compiled, not run on hardware
//
#include <hip/hip_runtime.h>

typedef _Float16 half_t;
typedef __attribute__((ext_vector_type(16))) _Float16 v16h;
typedef __attribute__((ext_vector_type(8)))  _Float16 v8h;
typedef __attribute__((ext_vector_type(8)))  float    v8f;

#define Bq 64
#define Nn 25
#define Cc 128
#define Tt 256
#define EPSV 1e-5f

// ---------------------------------------------------------------------------
// Kernel 0 (precompute, negligible cost):
//   Wcomb16[k][o][c] = (f16) sum_j W_conv[o][j][k] * W_gcn[j][c]
//   biasA[k][o]      = sum_j W_conv[o][j][k] * b_gcn[j]
//   rs[m]            = sum_n adj[m][n]      (bias was added before adj matmul)
// Folding the GCN linear into the conv weights removes a 13.4 GFLOP GEMM and
// a 105 MB intermediate (W_gcn acts on channels, adj on nodes: they commute).
// ---------------------------------------------------------------------------
__global__ __launch_bounds__(256) void k_prep(
    const float* __restrict__ Wg, const float* __restrict__ Wc,
    const float* __restrict__ bg, const float* __restrict__ adj,
    half_t* __restrict__ Wcomb16, float* __restrict__ biasA,
    float* __restrict__ rs) {
  int i = blockIdx.x * blockDim.x + threadIdx.x;   // over 3*128*128 = 49152
  if (i >= 3 * Cc * Cc) return;
  int k = i / (Cc * Cc);
  int rem = i - k * Cc * Cc;
  int o = rem >> 7, c = rem & 127;

  float acc = 0.f;
  for (int j = 0; j < Cc; ++j)
    acc += Wc[(o * Cc + j) * 3 + k] * Wg[j * Cc + c];
  Wcomb16[i] = (half_t)acc;

  if (c == 0) {                       // one thread per (k,o) does the bias dot
    float bb = 0.f;
    for (int j = 0; j < Cc; ++j) bb += Wc[(o * Cc + j) * 3 + k] * bg[j];
    biasA[k * Cc + o] = bb;
  }
  if (i < Nn) {                       // adjacency row sums
    float s = 0.f;
    for (int n = 0; n < Nn; ++n) s += adj[i * Nn + n];
    rs[i] = s;
  }
}

// ---------------------------------------------------------------------------
// Kernel 1: adjacency mix on the input (K=25 -> VALU), TRANSPOSED output:
//   x2t[b,m,t,c] = (f16) sum_n adj[m,n] * x[b,n,c,t]
// Accumulate in [c][t] order (coalesced x reads along t), transpose through
// LDS, write contiguous f16 rows. The [t][c] layout makes the consumer's LDS
// tile a single contiguous span -> async-to-LDS staging with no transpose.
// ---------------------------------------------------------------------------
__global__ __launch_bounds__(256) void k_adjmix(
    const float* __restrict__ x, const float* __restrict__ adj,
    half_t* __restrict__ x2t) {
  __shared__ __align__(16) half_t ldsT[16 * Cc];   // [t][c] tile, 4 KB

  const int tb = blockIdx.x & 15;                  // T/16 = 16 t-blocks
  const int bm = blockIdx.x >> 4;
  const int m = bm % Nn, b = bm / Nn;
  const int t0 = tb * 16;
  const int tid = threadIdx.x;
  const int c = tid >> 1;                          // 0..127
  const int th = (tid & 1) * 8;                    // t sub-offset 0 or 8

  const float* base = x + ((size_t)b * Nn * Cc + c) * Tt + t0 + th;

  float acc[8] = {0.f, 0.f, 0.f, 0.f, 0.f, 0.f, 0.f, 0.f};
#pragma unroll 5
  for (int nd = 0; nd < Nn; ++nd) {
    float a = adj[m * Nn + nd];
    v8f v = *(const v8f*)(base + (size_t)nd * Cc * Tt);
#pragma unroll
    for (int j = 0; j < 8; ++j) acc[j] += a * v[j];
  }
#pragma unroll
  for (int j = 0; j < 8; ++j)
    ldsT[(th + j) * Cc + c] = (half_t)acc[j];      // transpose into [t][c]
  __syncthreads();

  // contiguous coalesced write: tile image == global [t][c] rows
  half_t* po = x2t + ((size_t)(b * Nn + m) * Tt + t0) * Cc;
  *(v8h*)(po + tid * 8) = *(const v8h*)(&ldsT[tid * 8]);
}

// ---------------------------------------------------------------------------
// Kernel 2 (WMMA workhorse): for each (b, m, 32-wide t tile):
//   y = sum_k Wcomb_k(128x128) x2_shift_k(128x32)   (24 v_wmma per wave)
// Interior 32 rows of the LDS tile arrive via global_load_async_to_lds_b128
// (ASYNCcnt path, no VGPR round-trip); the 2 zero-padded halo rows use
// guarded plain loads. Fused epilogue: +bias(t,o)+b_conv -> BN1 -> ReLU
// -> +residual -> BN2 -> ReLU -> f32 out.
// ---------------------------------------------------------------------------
__global__ __launch_bounds__(256) void k_fused(
    const half_t* __restrict__ x2t, const half_t* __restrict__ Wcomb16,
    const float* __restrict__ biasA, const float* __restrict__ rs,
    const float* __restrict__ x, const float* __restrict__ b_conv,
    const float* __restrict__ tc_g, const float* __restrict__ tc_b,
    const float* __restrict__ tc_m, const float* __restrict__ tc_v,
    const float* __restrict__ bn_g, const float* __restrict__ bn_b,
    const float* __restrict__ bn_m, const float* __restrict__ bn_v,
    float* __restrict__ out) {
  __shared__ __align__(32) half_t ldsh[34 * Cc];   // rows: t0-1 .. t0+32

  const int tt = blockIdx.x & 7;
  const int bn = blockIdx.x >> 3;
  const int node = bn % Nn;
  const int t0 = tt * 32;
  const int tid = threadIdx.x;
  const half_t* hp = x2t + (size_t)bn * Tt * Cc;   // [t][c]

  // --- async copy: interior rows t0..t0+31 -> LDS rows 1..32 (8192 B) ---
  {
    const unsigned int lds_base =
        (unsigned int)(size_t)(void*)ldsh;           // low 32 bits = LDS addr
    const char* gsrc = (const char*)(hp + (size_t)t0 * Cc);
#pragma unroll
    for (int i = 0; i < 2; ++i) {
      int chunk = i * 256 + tid;                     // 0..511 x 16 B
      unsigned int la = lds_base + 256u + (unsigned int)chunk * 16u;
      unsigned long long ga = (unsigned long long)(size_t)(gsrc + chunk * 16);
      asm volatile("global_load_async_to_lds_b128 %0, %1, off"
                   :: "v"(la), "v"(ga)
                   : "memory");
    }
  }
  // --- halo rows (zero padding at t=-1 / t=T) via plain guarded path ---
  if (tid < 32) {
    int side = tid >> 4;                             // 0: t0-1 ; 1: t0+32
    int cq = (tid & 15) * 8;
    int gt = side ? (t0 + 32) : (t0 - 1);
    v8h v = {};
    if (gt >= 0 && gt < Tt) v = *(const v8h*)(hp + (size_t)gt * Cc + cq);
    *(v8h*)(&ldsh[(side ? 33 : 0) * Cc + cq]) = v;
  }

  const int wave = tid >> 5, lane = tid & 31;
  const int g = lane >> 4, n = lane & 15;
  const int mbase = wave * 16;
  const size_t base = (size_t)bn * Cc * Tt;

  // Warm residual rows for the epilogue (gfx1250 global_prefetch_b8 path).
  __builtin_prefetch(x + base + (size_t)(mbase + n) * Tt + t0, 0, 0);

  asm volatile("s_wait_asynccnt 0x0" ::: "memory");  // own async writes done
  __syncthreads();                                   // all waves' LDS visible

  v8f acc0 = {}, acc1 = {};
#pragma unroll
  for (int k = 0; k < 3; ++k) {
    const half_t* wk = Wcomb16 + k * Cc * Cc;
#pragma unroll
    for (int kb = 0; kb < 4; ++kb) {
      const int kbase = kb * 32;
      // A fragment: row o = mbase + lane%16; K contiguous 16g+e.
      const half_t* arow = wk + (mbase + n) * Cc + kbase;
      v8h alo = *(const v8h*)(arow + 16 * g);
      v8h ahi = *(const v8h*)(arow + 16 * g + 8);
      v16h a;
#pragma unroll
      for (int e = 0; e < 8; ++e) { a[e] = alo[e]; a[8 + e] = ahi[e]; }
      // B fragments: col t_local = sub*16 + n (+k tap shift), K contiguous.
      v16h b0 = *(const v16h*)(&ldsh[(n + k) * Cc + kbase + 16 * g]);
      v16h b1 = *(const v16h*)(&ldsh[(16 + n + k) * Cc + kbase + 16 * g]);
      acc0 = __builtin_amdgcn_wmma_f32_16x16x32_f16(
          false, a, false, b0, (short)0, acc0, false, false);
      acc1 = __builtin_amdgcn_wmma_f32_16x16x32_f16(
          false, a, false, b1, (short)0, acc1, false, false);
    }
  }

  const float rsm = rs[node];
#pragma unroll
  for (int r = 0; r < 8; ++r) {
    int o = mbase + r + 8 * g;       // C/D layout: M = r + 8*(lane/16)
    float s1 = tc_g[o] * rsqrtf(tc_v[o] + EPSV);
    float c1 = s1 * (b_conv[o] - tc_m[o]) + tc_b[o];
    float s2 = bn_g[o] * rsqrtf(bn_v[o] + EPSV);
    float c2 = bn_b[o] - s2 * bn_m[o];
    float ba0 = biasA[o], ba1 = biasA[Cc + o], ba2 = biasA[2 * Cc + o];
#pragma unroll
    for (int sub = 0; sub < 2; ++sub) {
      int t = t0 + sub * 16 + n;
      // b_gcn propagated through adj rowsum and conv taps; boundary taps see
      // the conv zero padding instead of the bias.
      float bias = rsm * (ba1 + (t > 0 ? ba0 : 0.f) + (t < Tt - 1 ? ba2 : 0.f));
      float v = (sub ? acc1[r] : acc0[r]) + bias;
      v = s1 * v + c1;                 // conv bias + BN1
      v = fmaxf(v, 0.f);               // ReLU
      v += x[base + o * Tt + t];       // residual
      v = s2 * v + c2;                 // BN2
      v = fmaxf(v, 0.f);               // ReLU
      out[base + o * Tt + t] = v;
    }
  }
}

// ---------------------------------------------------------------------------
extern "C" void kernel_launch(void* const* d_in, const int* in_sizes, int n_in,
                              void* d_out, int out_size, void* d_ws,
                              size_t ws_size, hipStream_t stream) {
  (void)in_sizes; (void)n_in; (void)out_size; (void)ws_size;
  const float* x    = (const float*)d_in[0];
  const float* adj  = (const float*)d_in[1];
  const float* Wg   = (const float*)d_in[2];
  const float* bg   = (const float*)d_in[3];
  const float* Wc   = (const float*)d_in[4];
  const float* bc   = (const float*)d_in[5];
  const float* tcg  = (const float*)d_in[6];
  const float* tcb  = (const float*)d_in[7];
  const float* tcm  = (const float*)d_in[8];
  const float* tcv  = (const float*)d_in[9];
  const float* bng  = (const float*)d_in[10];
  const float* bnb  = (const float*)d_in[11];
  const float* bnm  = (const float*)d_in[12];
  const float* bnv  = (const float*)d_in[13];
  float* out = (float*)d_out;

  const size_t n_elem = (size_t)Bq * Nn * Cc * Tt;   // 52,428,800
  char* ws = (char*)d_ws;
  half_t* x2t     = (half_t*)ws;                      // 105 MB f16 [b][m][t][c]
  half_t* Wcomb16 = (half_t*)(ws + n_elem * 2);       // 96 KB f16
  float*  biasA   = (float*)(ws + n_elem * 2 + 3 * Cc * Cc * 2);  // 1.5 KB
  float*  rsum    = biasA + 3 * Cc;                   // 100 B

  k_prep<<<(3 * Cc * Cc + 255) / 256, 256, 0, stream>>>(Wg, Wc, bg, adj,
                                                        Wcomb16, biasA, rsum);
  k_adjmix<<<Bq * Nn * 16, 256, 0, stream>>>(x, adj, x2t);
  k_fused<<<Bq * Nn * 8, 256, 0, stream>>>(x2t, Wcomb16, biasA, rsum, x, bc,
                                           tcg, tcb, tcm, tcv,
                                           bng, bnb, bnm, bnv, out);
}
